// NeuraMatch_90984587199166
// MI455X (gfx1250) — matrix operand order repaired and checked by reading, not compiled, not run on hardware
//
#include <hip/hip_runtime.h>
#include <hip/hip_bf16.h>
#include <cstdint>

// ---------------------------------------------------------------------------
// MI455X (gfx1250) implementation.
//  * Convs: implicit GEMM with v_wmma_f32_16x16x32_f16, f16 activations (NHWC)
//    staged in LDS, f32 accumulators, fused BN + LeakyReLU epilogue.
//  * Each wave computes 32 output pixels (2 A-frags) so every B fragment is
//    reused by two WMMAs -> ~1.5 memory ops per WMMA in the K loop.
//  * Weights pre-packed on device into exact WMMA B-fragment lane order.
//  * f16 keeps the biggest intermediate (131MB) inside the 192MB L2.
//  * Top-20 selection with stable-argsort tie-break semantics; tiny MLP matcher.
// ---------------------------------------------------------------------------

typedef _Float16 h16;
typedef __attribute__((ext_vector_type(16))) _Float16 v16h;
typedef __attribute__((ext_vector_type(8)))  float    v8f;

union AFrag { v16h h; uint4 u[2]; };
static_assert(sizeof(v16h) == 32, "v16h must be 32 bytes");

#define NEGINF (-3.402823466e38f)

// ------------------------- input packing (f32 NCHW -> f16 NHWC) -------------
__global__ __launch_bounds__(256)
void pack_inputs_kernel(const float* __restrict__ xa, const float* __restrict__ xb,
                        h16* __restrict__ xa16, h16* __restrict__ xb16,
                        h16* __restrict__ xab16) {
    const int HW = 512, TOT = 2 * HW * HW;
    int i = blockIdx.x * 256 + threadIdx.x;
    if (i >= TOT) return;
    int b = i / (HW * HW);
    int px = i - b * HW * HW;
    for (int c = 0; c < 3; ++c) {
        float va = xa[((size_t)(b * 3 + c)) * HW * HW + px];
        float vb = xb[((size_t)(b * 3 + c)) * HW * HW + px];
        xa16[(size_t)i * 3 + c] = (h16)va;
        xb16[(size_t)i * 3 + c] = (h16)vb;
        xab16[(size_t)i * 6 + c] = (h16)va;
        xab16[(size_t)i * 6 + 3 + c] = (h16)vb;
    }
}

// ------------------- weight pack into WMMA B-fragment order -----------------
// k ordering: k = (ky*KZ + kx)*CI + ci.  B 32x16 layout (wave32):
// lane 0-15: N=lane, K=0..7,16..23 ; lane 16-31: N=lane-16, K=8..15,24..31.
// pack index t = ((nt*KB + kb)*32 + lane)*16 + j  ->  K = kb*32 + hi*8 + (j&7) + (j>>3)*16.
__global__ __launch_bounds__(256)
void pack_weights_kernel(const float* __restrict__ w, const float* __restrict__ g,
                         const float* __restrict__ be, const float* __restrict__ mu,
                         const float* __restrict__ var,
                         h16* __restrict__ wpack, float2* __restrict__ scb,
                         int CO, int CI, int KZ) {
    const int Ktot = CI * KZ * KZ;
    const int KB = (Ktot + 31) / 32;
    const int NT = CO / 16;
    const int total = NT * KB * 512;
    int t = blockIdx.x * 256 + threadIdx.x;
    if (t < CO) {
        float sc = g[t] * rsqrtf(var[t] + 1e-5f);
        scb[t] = make_float2(sc, be[t] - mu[t] * sc);
    }
    if (t >= total) return;
    int j    = t & 15;
    int lane = (t >> 4) & 31;
    int rest = t >> 9;
    int kb = rest % KB;
    int nt = rest / KB;
    int n  = lane & 15;
    int hi = lane >> 4;
    int k  = kb * 32 + hi * 8 + (j & 7) + (j >> 3) * 16;
    int co = nt * 16 + n;
    float val = 0.f;
    if (k < Ktot) {
        int ci = k % CI;
        int kk = k / CI;
        int ky = kk / KZ, kx = kk % KZ;
        val = w[(((size_t)co * CI + ci) * KZ + ky) * KZ + kx];
    }
    wpack[t] = (h16)val;
}

// ----------------------------- conv (implicit GEMM) -------------------------
// Block: 64 threads = 2 wave32s. Each wave owns 32 consecutive output pixels
// (two 16-row A fragments), computes all CO channels as CO/16 WMMA N-tiles.
// Each B fragment load feeds two WMMAs.  Input patch KZ x (64+KZ-1) x CI is
// staged once in LDS (<= 50.7 KB, well under the 320 KB WGP budget).
template <int CI, int CO, int KZ>
__global__ void __launch_bounds__(64)
conv_wmma_kernel(const h16* __restrict__ in, const h16* __restrict__ wpack,
                 const float2* __restrict__ scb, h16* __restrict__ out,
                 int HI, int WI) {
    constexpr int TW   = 64 + KZ - 1;
    constexpr int Ktot = CI * KZ * KZ;
    constexpr int KB   = (Ktot + 31) / 32;
    constexpr int NT   = CO / 16;
    const int HO = HI - KZ + 1;
    const int WO = WI - KZ + 1;
    const int ox0 = blockIdx.x * 64;
    const int oy  = blockIdx.y;
    const int b   = blockIdx.z;
    const int tid = threadIdx.x;
    const int lane = tid & 31;
    const int wv   = tid >> 5;          // 0 or 1

    __shared__ __align__(16) h16 tile[KZ][TW][CI];

    // ---- stage input patch into LDS ----
    const h16* inb = in + ((size_t)(b * HI + oy) * WI) * CI;
    if constexpr (CI % 8 == 0) {
        constexpr int VC = CI / 8;
        for (int f = tid; f < KZ * TW * VC; f += 64) {
            int c8 = f % VC;
            int rest = f / VC;
            int x  = rest % TW;
            int ky = rest / TW;
            int gx = ox0 + x;
            uint4 val = make_uint4(0u, 0u, 0u, 0u);
            if (gx < WI) val = *(const uint4*)(inb + ((size_t)ky * WI + gx) * CI + c8 * 8);
            *(uint4*)&tile[ky][x][c8 * 8] = val;
        }
    } else {
        for (int f = tid; f < KZ * TW * CI; f += 64) {
            int c = f % CI;
            int rest = f / CI;
            int x  = rest % TW;
            int ky = rest / TW;
            int gx = ox0 + x;
            h16 v = (h16)0.f;
            if (gx < WI) v = inb[((size_t)ky * WI + gx) * CI + c];
            tile[ky][x][c] = v;
        }
    }
    __syncthreads();

    const int mrow = lane & 15;
    const int hi   = lane >> 4;
    const int m0   = wv * 32 + mrow;     // A rows for fragment 0
    const int m1   = m0 + 16;            // A rows for fragment 1

    v8f acc0[NT], acc1[NT];
#pragma unroll
    for (int t = 0; t < NT; ++t)
#pragma unroll
        for (int e = 0; e < 8; ++e) { acc0[t][e] = 0.f; acc1[t][e] = 0.f; }

    for (int kb = 0; kb < KB; ++kb) {
        AFrag a0, a1;
        if constexpr (CI % 32 == 0) {
            // one 32-run stays inside a single (ky,kx): two b128 LDS loads each.
            const int k0 = kb * 32;
            const int kk = k0 / CI;
            const int c0 = k0 % CI;
            const int ky = kk / KZ, kx = kk % KZ;
            const h16* p0 = &tile[ky][m0 + kx][c0 + hi * 8];
            const h16* p1 = &tile[ky][m1 + kx][c0 + hi * 8];
            a0.u[0] = *(const uint4*)p0;        // K {0..7}  (+8 for hi lanes)
            a0.u[1] = *(const uint4*)(p0 + 16); // K {16..23}(+8 for hi lanes)
            a1.u[0] = *(const uint4*)p1;
            a1.u[1] = *(const uint4*)(p1 + 16);
        } else {
#pragma unroll
            for (int j = 0; j < 16; ++j) {
                int off = j + hi * 8 + (j >= 8 ? 8 : 0);
                int k = kb * 32 + off;
                h16 v0 = (h16)0.f, v1 = (h16)0.f;
                if (k < Ktot) {
                    int c  = k % CI;
                    int kk = k / CI;
                    int ky = kk / KZ, kx = kk % KZ;
                    v0 = tile[ky][m0 + kx][c];
                    v1 = tile[ky][m1 + kx][c];
                }
                a0.h[j] = v0;
                a1.h[j] = v1;
            }
        }
#pragma unroll
        for (int t = 0; t < NT; ++t) {
            const v16h* bp =
                (const v16h*)(wpack + (((size_t)t * KB + kb) * 32 + lane) * 16);
            v16h bfrag = *bp;                    // one B load, two WMMAs
            acc0[t] = __builtin_amdgcn_wmma_f32_16x16x32_f16(
                false, a0.h, false, bfrag, (short)0, acc0[t], false, false);
            acc1[t] = __builtin_amdgcn_wmma_f32_16x16x32_f16(
                false, a1.h, false, bfrag, (short)0, acc1[t], false, false);
        }
    }

    // ---- epilogue: BN + LeakyReLU(0.01), store f16 NHWC ----
#pragma unroll
    for (int t = 0; t < NT; ++t) {
        int co = t * 16 + mrow;           // N index = lane&15
        float2 sb = scb[co];
#pragma unroll
        for (int r2 = 0; r2 < 8; ++r2) {
            int mo = wv * 32 + r2 + hi * 8;   // M index, fragment 0
            int ox = ox0 + mo;
            if (ox < WO) {
                float v = acc0[t][r2] * sb.x + sb.y;
                v = v > 0.f ? v : 0.01f * v;
                out[((size_t)(b * HO + oy) * WO + ox) * CO + co] = (h16)v;
            }
            int ox1 = ox + 16;                 // fragment 1
            if (ox1 < WO) {
                float v = acc1[t][r2] * sb.x + sb.y;
                v = v > 0.f ? v : 0.01f * v;
                out[((size_t)(b * HO + oy) * WO + ox1) * CO + co] = (h16)v;
            }
        }
    }
}

// ---------------------- final 32->1 conv + sigmoid (f32) --------------------
__global__ __launch_bounds__(128)
void final_conv_kernel(const h16* __restrict__ h, const float* __restrict__ w,
                       const float* __restrict__ bias, float* __restrict__ heat) {
    const int S = 502, HIN = 504;
    int ox = blockIdx.x * 128 + threadIdx.x;
    int oy = blockIdx.y;
    int b  = blockIdx.z;
    if (ox >= S) return;
    float acc = bias[0];
    for (int ky = 0; ky < 3; ++ky)
        for (int kx = 0; kx < 3; ++kx) {
            const h16* p = h + ((size_t)(b * HIN + oy + ky) * HIN + ox + kx) * 32;
#pragma unroll
            for (int c = 0; c < 32; ++c)
                acc += (float)p[c] * w[(c * 3 + ky) * 3 + kx];
        }
    heat[(size_t)b * S * S + oy * S + ox] = 1.f / (1.f + expf(-acc));
}

// ----------------------------- top-20 selection -----------------------------
// Comparator replicates stable jnp.argsort ascending + take-last-K:
// "greater" = (v1>v2) || (v1==v2 && i1>i2).
__device__ __forceinline__ bool tk_gt(float v1, int i1, float v2, int i2) {
    return (v1 > v2) || (v1 == v2 && i1 > i2);
}

__global__ __launch_bounds__(256)
void topk_kernel(const float* __restrict__ heat, const float* __restrict__ thr_p,
                 int* __restrict__ tkidx) {
    const int S2 = 252004;
    const int blk = blockIdx.x;          // (b<<1)|neg
    const int b = blk >> 1, neg = blk & 1;
    const float thr = thr_p[0];
    const float* hm = heat + (size_t)b * S2;

    float lv[20]; int li[20];
#pragma unroll
    for (int q = 0; q < 20; ++q) { lv[q] = NEGINF; li[q] = -1; }

    for (int i = threadIdx.x; i < S2; i += 256) {
        float v = hm[i];
        bool m = v > thr;
        if (neg) m = !m;
        float key = m ? v : NEGINF;
        if (tk_gt(key, i, lv[19], li[19])) {
            int p = 19;
            while (p > 0 && tk_gt(key, i, lv[p - 1], li[p - 1])) {
                lv[p] = lv[p - 1]; li[p] = li[p - 1]; --p;
            }
            lv[p] = key; li[p] = i;
        }
    }

    __shared__ float sv[256 * 20];
    __shared__ int   si[256 * 20];
    for (int q = 0; q < 20; ++q) { sv[threadIdx.x * 20 + q] = lv[q]; si[threadIdx.x * 20 + q] = li[q]; }
    __syncthreads();

    // level 1: 32 threads merge 8 lists each
    float mv[20]; int mi[20];
    if (threadIdx.x < 32) {
#pragma unroll
        for (int q = 0; q < 20; ++q) { mv[q] = NEGINF; mi[q] = INT_MIN; }
        int base = threadIdx.x * 160;
        for (int e = 0; e < 160; ++e) {
            float v = sv[base + e]; int ii = si[base + e];
            if (tk_gt(v, ii, mv[19], mi[19])) {
                int p = 19;
                while (p > 0 && tk_gt(v, ii, mv[p - 1], mi[p - 1])) {
                    mv[p] = mv[p - 1]; mi[p] = mi[p - 1]; --p;
                }
                mv[p] = v; mi[p] = ii;
            }
        }
    }
    __syncthreads();
    if (threadIdx.x < 32) {
        for (int q = 0; q < 20; ++q) { sv[threadIdx.x * 20 + q] = mv[q]; si[threadIdx.x * 20 + q] = mi[q]; }
    }
    __syncthreads();

    // level 2: thread 0 merges 32 lists; result sorted descending.
    if (threadIdx.x == 0) {
        float fv[20]; int fi[20];
#pragma unroll
        for (int q = 0; q < 20; ++q) { fv[q] = NEGINF; fi[q] = INT_MIN; }
        for (int e = 0; e < 640; ++e) {
            float v = sv[e]; int ii = si[e];
            if (tk_gt(v, ii, fv[19], fi[19])) {
                int p = 19;
                while (p > 0 && tk_gt(v, ii, fv[p - 1], fi[p - 1])) {
                    fv[p] = fv[p - 1]; fi[p] = fi[p - 1]; --p;
                }
                fv[p] = v; fi[p] = ii;
            }
        }
        // reference order: ascending heatmap value (argsort[-K:])
        for (int r = 0; r < 20; ++r) tkidx[blk * 20 + (19 - r)] = fi[r];
    }
}

// --------------------------------- matcher ----------------------------------
__global__ __launch_bounds__(192)
void matcher_kernel(const float* __restrict__ heat, const h16* __restrict__ fa,
                    const h16* __restrict__ fb, const int* __restrict__ tkidx,
                    const float* __restrict__ W1, const float* __restrict__ g1,
                    const float* __restrict__ be1, const float* __restrict__ mu1,
                    const float* __restrict__ va1,
                    const float* __restrict__ W2, const float* __restrict__ g2,
                    const float* __restrict__ be2, const float* __restrict__ mu2,
                    const float* __restrict__ va2,
                    const float* __restrict__ W3, const float* __restrict__ b3,
                    const float* __restrict__ fthr_p, float* __restrict__ o) {
    const int S2 = 252004, S = 502, NP = 190;
    const int blk = blockIdx.x;
    const int b = blk >> 1, neg = blk & 1;
    int p = threadIdx.x;
    if (p >= NP) return;

    // triu indices (lexicographic i<j, K=20)
    int i = 0, rem = p;
    while (rem >= 19 - i) { rem -= 19 - i; ++i; }
    int j = i + 1 + rem;

    const int* tk = tkidx + blk * 20;
    int ia = tk[i], ib = tk[j];

    const float r = 512.0f / 502.0f;
    float* pp = o + 504008 + (size_t)neg * 26600; // p   (NB,190,4)
    float* yy = pp + 1520;                        // y   (NB,190)
    float* ff = yy + 380;                         // fab (NB,190,64)
    float* ss = ff + 24320;                       // sel (NB,190)

    size_t po = ((size_t)b * NP + p) * 4;
    pp[po + 0] = (float)(ia % S) * r;
    pp[po + 1] = (float)(ia / S) * r;
    pp[po + 2] = (float)(ib % S) * r;
    pp[po + 3] = (float)(ib / S) * r;

    float fab[64];
    const h16* fpa = fa + ((size_t)b * S2 + ia) * 32;
    const h16* fpb = fb + ((size_t)b * S2 + ib) * 32;
#pragma unroll
    for (int c = 0; c < 32; ++c) { fab[c] = (float)fpa[c]; fab[32 + c] = (float)fpb[c]; }
    float* fo = ff + ((size_t)b * NP + p) * 64;
    for (int c = 0; c < 64; ++c) fo[c] = fab[c];

    float h1[32];
    for (int n = 0; n < 32; ++n) {
        float s = 0.f;
        for (int c = 0; c < 64; ++c) s += W1[n * 64 + c] * fab[c];
        float sc = g1[n] * rsqrtf(va1[n] + 1e-5f);
        h1[n] = (s - mu1[n]) * sc + be1[n];
    }
    float h2[16];
    for (int n = 0; n < 16; ++n) {
        float s = 0.f;
        for (int c = 0; c < 32; ++c) s += W2[n * 32 + c] * h1[c];
        float sc = g2[n] * rsqrtf(va2[n] + 1e-5f);
        h2[n] = (s - mu2[n]) * sc + be2[n];
    }
    float s3 = b3[0];
    for (int c = 0; c < 16; ++c) s3 += W3[c] * h2[c];
    float y = 1.f / (1.f + expf(-s3));

    const float* hm = heat + (size_t)b * S2;
    float ys = (y + hm[ia] + hm[ib]) * (1.0f / 3.0f);
    yy[(size_t)b * NP + p] = ys;
    ss[(size_t)b * NP + p] = (ys > fthr_p[0]) ? 1.f : 0.f;
}

// --------------------------------- launch -----------------------------------
extern "C" void kernel_launch(void* const* d_in, const int* in_sizes, int n_in,
                              void* d_out, int out_size, void* d_ws, size_t ws_size,
                              hipStream_t stream) {
    (void)in_sizes; (void)n_in; (void)out_size; (void)ws_size;
    const int NB = 2, HW = 512;

    char* ws = (char*)d_ws;
    size_t off = 0;
    auto alloc = [&](size_t bytes) -> void* {
        void* p = ws + off;
        off += (bytes + 255) & ~(size_t)255;
        return p;
    };

    h16* xa16  = (h16*)alloc((size_t)NB * HW * HW * 3 * 2);
    h16* xb16  = (h16*)alloc((size_t)NB * HW * HW * 3 * 2);
    h16* xab16 = (h16*)alloc((size_t)NB * HW * HW * 6 * 2);
    h16* T1 = (h16*)alloc((size_t)NB * 510 * 510 * 64 * 2);   // 66.6 MB
    h16* T2 = (h16*)alloc((size_t)NB * 508 * 508 * 64 * 2);   // 66.1 MB
    h16* T3 = (h16*)alloc((size_t)NB * 506 * 506 * 128 * 2);  // 131 MB (fits L2)
    h16* fa = (h16*)alloc((size_t)NB * 502 * 502 * 32 * 2);
    h16* fb = (h16*)alloc((size_t)NB * 502 * 502 * 32 * 2);

    auto wp_halves = [](int CO, int CI, int KZ) {
        return (size_t)(CO / 16) * ((CI * KZ * KZ + 31) / 32) * 512;
    };
    h16*    wp[10]; float2* sb[10];
    const int shapes[10][3] = { {64,3,6},{64,64,4},{32,64,3},      // block_a
                                {64,3,6},{64,64,4},{32,64,3},      // block_b
                                {64,6,3},{64,64,3},{128,64,3},{32,128,3} }; // block_ab
    for (int l = 0; l < 10; ++l) {
        wp[l] = (h16*)alloc(wp_halves(shapes[l][0], shapes[l][1], shapes[l][2]) * 2);
        sb[l] = (float2*)alloc((size_t)shapes[l][0] * sizeof(float2));
    }
    int* tkidx = (int*)alloc(4 * 20 * sizeof(int));

    const float* x_a = (const float*)d_in[0];
    const float* x_b = (const float*)d_in[1];
    // params flattened in setup_inputs() dict insertion order:
    //  block_a  L0: w=2 bn=3..6   L1: 7,8..11   L2: 12,13..16
    //  block_b  L0: 17,18..21     L1: 22,23..26 L2: 27,28..31
    //  block_ab L0: 32,..36  L1: 37,..41  L2: 42,..46  L3: 47,..51
    //  ab_final_w=52 ab_final_b=53
    //  matcher: W1=54 bn1=55..58 W2=59 bn2=60..63 W3=64 b3=65
    //  heatmap_thresh=66 final_thresh=67
    const int widx[10] = {2, 7, 12, 17, 22, 27, 32, 37, 42, 47};

    pack_inputs_kernel<<<(NB * HW * HW + 255) / 256, 256, 0, stream>>>(
        x_a, x_b, xa16, xb16, xab16);

    for (int l = 0; l < 10; ++l) {
        int CO = shapes[l][0], CI = shapes[l][1], KZ = shapes[l][2];
        size_t total = wp_halves(CO, CI, KZ);
        size_t n = total > (size_t)CO ? total : (size_t)CO;
        pack_weights_kernel<<<(int)((n + 255) / 256), 256, 0, stream>>>(
            (const float*)d_in[widx[l]], (const float*)d_in[widx[l] + 1],
            (const float*)d_in[widx[l] + 2], (const float*)d_in[widx[l] + 3],
            (const float*)d_in[widx[l] + 4], wp[l], sb[l], CO, CI, KZ);
    }

    // block_a : 512 -> 507 -> 504 -> 502
    conv_wmma_kernel<3, 64, 6><<<dim3(8, 507, NB), 64, 0, stream>>>(xa16, wp[0], sb[0], T1, 512, 512);
    conv_wmma_kernel<64, 64, 4><<<dim3(8, 504, NB), 64, 0, stream>>>(T1, wp[1], sb[1], T2, 507, 507);
    conv_wmma_kernel<64, 32, 3><<<dim3(8, 502, NB), 64, 0, stream>>>(T2, wp[2], sb[2], fa, 504, 504);
    // block_b
    conv_wmma_kernel<3, 64, 6><<<dim3(8, 507, NB), 64, 0, stream>>>(xb16, wp[3], sb[3], T1, 512, 512);
    conv_wmma_kernel<64, 64, 4><<<dim3(8, 504, NB), 64, 0, stream>>>(T1, wp[4], sb[4], T2, 507, 507);
    conv_wmma_kernel<64, 32, 3><<<dim3(8, 502, NB), 64, 0, stream>>>(T2, wp[5], sb[5], fb, 504, 504);
    // block_ab : 512 -> 510 -> 508 -> 506 -> 504
    conv_wmma_kernel<6, 64, 3><<<dim3(8, 510, NB), 64, 0, stream>>>(xab16, wp[6], sb[6], T1, 512, 512);
    conv_wmma_kernel<64, 64, 3><<<dim3(8, 508, NB), 64, 0, stream>>>(T1, wp[7], sb[7], T2, 510, 510);
    conv_wmma_kernel<64, 128, 3><<<dim3(8, 506, NB), 64, 0, stream>>>(T2, wp[8], sb[8], T3, 508, 508);
    conv_wmma_kernel<128, 32, 3><<<dim3(8, 504, NB), 64, 0, stream>>>(T3, wp[9], sb[9], T1, 506, 506);

    float* heat = (float*)d_out;
    final_conv_kernel<<<dim3(4, 502, NB), 128, 0, stream>>>(
        T1, (const float*)d_in[52], (const float*)d_in[53], heat);

    topk_kernel<<<4, 256, 0, stream>>>(heat, (const float*)d_in[66], tkidx);

    matcher_kernel<<<4, 192, 0, stream>>>(
        heat, fa, fb, tkidx,
        (const float*)d_in[54], (const float*)d_in[55], (const float*)d_in[56],
        (const float*)d_in[57], (const float*)d_in[58],
        (const float*)d_in[59], (const float*)d_in[60], (const float*)d_in[61],
        (const float*)d_in[62], (const float*)d_in[63],
        (const float*)d_in[64], (const float*)d_in[65],
        (const float*)d_in[67], (float*)d_out);
}
